// Encoder_Processor_70265664962700
// MI455X (gfx1250) — matrix operand order repaired and checked by reading, not compiled, not run on hardware
//
#include <hip/hip_runtime.h>
#include <hip/hip_bf16.h>

// Problem constants (from reference setup_inputs)
#define S_LEN   256
#define BATCH   64
#define IN_DIM  256
#define H_DIM   256
#define INNER   512
#define A_DIM   64
#define NLEVELS 7        // DEPTH - 1
#define LEAKC   0.01f

typedef __attribute__((ext_vector_type(16))) _Float16 v16h;
typedef __attribute__((ext_vector_type(8)))  float    v8f;

__device__ __forceinline__ v8f wmma_f16f32(v16h a, v16h b, v8f c) {
#if defined(__HIP_DEVICE_COMPILE__)
  // D = A(16x32 f16) * B(32x16 f16) + C(16x16 f32)
  return __builtin_amdgcn_wmma_f32_16x16x32_f16(false, a, false, b, (short)0, c,
                                                false, false);
#else
  return c;
#endif
}

__device__ __forceinline__ float lrelu(float v) { return v >= 0.0f ? v : LEAKC * v; }

// -------- packed WMMA operand layouts (CDNA5 ISA §7.12.2) --------
// A 16x32 f16 tile, per-lane 16 halves:
//   lane<16 : M=lane,    K = {0..7, 16..23}
//   lane>=16: M=lane-16, K = {8..15, 24..31}
// B 32x16 f16 tile, per-lane 16 halves:
//   lane<16 : N=lane,    K = {0..7, 16..23}
//   lane>=16: N=lane-16, K = {8..15, 24..31}
// Packed storage: panel-major, 32 lanes x 16 halves contiguous (32B/lane).

__device__ __forceinline__ v16h ld_pack(const _Float16* base, int panel, int lane) {
  return *(const v16h*)(base + (((size_t)panel * 32 + lane) << 4));
}

// flat offset (in halves) of element (m, kk) inside packed-A panel grid
__device__ __forceinline__ int a_pack_idx(int mp, int kPanels, int kp, int m, int kk) {
  int lane = m + (((kk >> 3) & 1) << 4);
  int idx = (kk & 7) + ((kk >> 4) << 3);
  return ((mp * kPanels + kp) * 32 + lane) * 16 + idx;
}

// ---------------- weight pack: f32 row-major [K,N] -> packed-B f16 ----------------
__global__ void pack_b(const float* __restrict__ B, _Float16* __restrict__ P, int K, int N) {
  int o = blockIdx.x * blockDim.x + threadIdx.x;
  if (o >= K * N) return;
  int idx = o & 15, lane = (o >> 4) & 31, panel = o >> 9;
  int tilesN = N >> 4;
  int np = panel % tilesN, kp = panel / tilesN;
  int n = np * 16 + (lane & 15);
  int k = kp * 32 + ((lane < 16) ? 0 : 8) + (idx & 7) + ((idx >> 3) << 4);
  P[o] = (_Float16)B[(size_t)k * N + n];
}

// ---------------- batched (time-parallel) GEMM ----------------
// C[M,N] = A[M,K](f32) @ Bp[K,N](packed f16) + bias[N].
// 8 waves/block; each wave computes a 16x64 supertile: one A load feeds 4 WMMA.
// All 4 B-tiles are loaded into distinct registers first (one long load clause),
// then 4 independent WMMAs issue back-to-back.
// remap!=0: output row r=(s*BATCH+b) reads A row (b*S_LEN+s)  (embedding transpose)
__global__ __launch_bounds__(256) void gemm_wmma(
    const float* __restrict__ A, int lda, int remap,
    const _Float16* __restrict__ Bp, const float* __restrict__ bias,
    float* __restrict__ C, int M, int N, int K) {
  int wave = threadIdx.x >> 5, lane = threadIdx.x & 31;
  int tilesN = N >> 4;       // 16-wide tiles
  int tiles4 = N >> 6;       // 64-wide supertiles
  int tile = blockIdx.x * 8 + wave;
  if (tile >= (M >> 4) * tiles4) return;
  int tm = tile / tiles4, tn4 = tile - tm * tiles4;

  int row = tm * 16 + (lane & 15);
  const float* arow;
  if (remap) {
    int s = row >> 6, bb = row & 63;                 // BATCH = 64
    arow = A + ((size_t)bb * S_LEN + s) * lda;
  } else {
    arow = A + (size_t)row * lda;
  }
  int kb = (lane < 16) ? 0 : 8;

  v8f acc0 = {}, acc1 = {}, acc2 = {}, acc3 = {};
  for (int k0 = 0; k0 < K; k0 += 32) {
    const float* p = arow + k0 + kb;
    int kp = k0 >> 5;
    const _Float16* pb = Bp + ((((size_t)kp * tilesN + tn4 * 4) * 32 + lane) << 4);
    __builtin_prefetch(pb + (size_t)tilesN * 512, 0, 1);   // next K-panel

    // issue all operand loads first: 4 f32 b128 (A) + 8 f16 b128 (B) in one clause
    v16h b0 = *(const v16h*)(pb);
    v16h b1 = *(const v16h*)(pb + 512);
    v16h b2 = *(const v16h*)(pb + 1024);
    v16h b3 = *(const v16h*)(pb + 1536);
    v16h av;
#pragma unroll
    for (int i = 0; i < 8; ++i) { av[i] = (_Float16)p[i]; av[8 + i] = (_Float16)p[16 + i]; }

    acc0 = wmma_f16f32(av, b0, acc0);
    acc1 = wmma_f16f32(av, b1, acc1);
    acc2 = wmma_f16f32(av, b2, acc2);
    acc3 = wmma_f16f32(av, b3, acc3);
  }
  int mo = (lane < 16) ? 0 : 8;
  v8f accs[4] = {acc0, acc1, acc2, acc3};
#pragma unroll
  for (int j = 0; j < 4; ++j) {
    int col = (tn4 * 4 + j) * 16 + (lane & 15);
    float bb = bias[col];
#pragma unroll
    for (int r2 = 0; r2 < 8; ++r2) {
      int orow = tm * 16 + r2 + mo;
      C[(size_t)orow * N + col] = accs[j][r2] + bb;
    }
  }
}

// ---------------- sequential scan for one vertical level ----------------
// One persistent workgroup: 1024 threads = 32 wave32 waves; recurrent state in LDS.
// LDS: h_l f32[64][256] | a_l f32[64][64] | gates 4xf32[64]
//      | hA packed-A f16 of h (4 mp x 8 kp panels)   | hsA packed-A f16 of h_ (4 x 16)
#define SCAN_LDS_BYTES (64*256*4 + 64*64*4 + 4*64*4 + 64*256*2 + 64*512*2)

__global__ __launch_bounds__(1024, 1) void scan_level(
    const float* __restrict__ XE,        // [S,B,H] level input (f32)
    const float* __restrict__ XW,        // [S,B,INNER] = xt@W + b
    const float* __restrict__ XA,        // [S,B,A]     = xt@W_action_1 + b_action_1
    const _Float16* __restrict__ Up,     // packed [H,INNER]
    const _Float16* __restrict__ W1p,    // packed [INNER,H]
    const _Float16* __restrict__ UA1p,   // packed [H,A]
    const float* __restrict__ b1,        // [H]
    const float* __restrict__ WA3,       // [A]
    const float* __restrict__ bA3,       // [1]
    const float* __restrict__ maskBS,    // [B,S]
    const float* __restrict__ NMSp,      // [S,B] from previous level
    const float* __restrict__ HVSp,      // [S,B]
    float* __restrict__ NMSc, float* __restrict__ HVSc,
    float* __restrict__ HS,              // [S,B,H] next-level input
    float* __restrict__ OUT,             // [B,H]
    int level, int lastLevel) {
  extern __shared__ char smem[];
  float*    h_l    = (float*)smem;                 // 64*256 f32
  float*    a_l    = h_l + 64 * 256;               // 64*64 f32
  float*    both_l = a_l + 64 * 64;                // 64
  float*    xo_l   = both_l + 64;                  // 64
  float*    ho_l   = xo_l + 64;                    // 64
  float*    hv_l   = ho_l + 64;                    // 64
  _Float16* hA     = (_Float16*)(hv_l + 64);       // 64*256 f16 packed-A (kPanels=8)
  _Float16* hsA    = hA + 64 * 256;                // 64*512 f16 packed-A (kPanels=16)

  const int tid = threadIdx.x;
  const int wave = tid >> 5, lane = tid & 31;
  const int lm = lane & 15;
  const int mo = (lane < 16) ? 0 : 8;

  for (int i = tid; i < 64 * 256; i += 1024) { h_l[i] = 0.0f; hA[i] = (_Float16)0.0f; }
  if (tid < 64) hv_l[tid] = 0.0f;
  __syncthreads();

  for (int t = 0; t < S_LEN; ++t) {
    const float* XE_t = XE + (size_t)t * BATCH * H_DIM;
    const float* XW_t = XW + (size_t)t * BATCH * INNER;
    const float* XA_t = XA + (size_t)t * BATCH * A_DIM;

    // ---- Lookahead: pull next timestep's streams toward WGP$ while WMMAs run.
    // XW_{t+1}: 128KB = 1024 x 128B lines (one per thread); XE_{t+1}: 512 lines;
    // XA_{t+1}: 128 lines. Counter-free global_prefetch_b8.
    if (t + 1 < S_LEN) {
      __builtin_prefetch(XW + (size_t)(t + 1) * BATCH * INNER + (size_t)tid * 32, 0, 1);
      if (tid < 512)
        __builtin_prefetch(XE + (size_t)(t + 1) * BATCH * H_DIM + (size_t)tid * 32, 0, 1);
      if (tid < 128)
        __builtin_prefetch(XA + (size_t)(t + 1) * BATCH * A_DIM + (size_t)tid * 32, 0, 1);
    }

    // ---- Phase 1: h_ = lrelu(XW_t + h@U)  and  a = lrelu(XA_t + h@U_action_1) ----
    // 128 HU tiles (4x32) + 16 ACT tiles (4x4) across 32 waves.
    for (int tt = wave; tt < 144; tt += 32) {
      if (tt < 128) {
        int tm = tt >> 5, tn = tt & 31;
        v8f acc = {};
#pragma unroll
        for (int kp = 0; kp < 8; ++kp) {
          v16h av = ld_pack(hA, tm * 8 + kp, lane);           // ds_load_b128 x2
          v16h bv = ld_pack(Up, kp * 32 + tn, lane);          // global b128 x2
          acc = wmma_f16f32(av, bv, acc);
        }
        int col = tn * 16 + lm;
        int kp = col >> 5, kk = col & 31;
#pragma unroll
        for (int r2 = 0; r2 < 8; ++r2) {
          int m = r2 + mo;
          float v = lrelu(acc[r2] + XW_t[(size_t)(tm * 16 + m) * INNER + col]);
          hsA[a_pack_idx(tm, 16, kp, m, kk)] = (_Float16)v;   // ds_store_b16
        }
      } else {
        int q = tt - 128;
        int tm = q >> 2, tn = q & 3;
        v8f acc = {};
#pragma unroll
        for (int kp = 0; kp < 8; ++kp) {
          v16h av = ld_pack(hA, tm * 8 + kp, lane);
          v16h bv = ld_pack(UA1p, kp * 4 + tn, lane);
          acc = wmma_f16f32(av, bv, acc);
        }
        int col = tn * 16 + lm;
#pragma unroll
        for (int r2 = 0; r2 < 8; ++r2) {
          int row = tm * 16 + r2 + mo;
          a_l[row * A_DIM + col] = lrelu(acc[r2] + XA_t[(size_t)row * A_DIM + col]);
        }
      }
    }
    __syncthreads();

    // ---- Phase 2: per-sample gate math (one thread per batch row) ----
    if (tid < BATCH) {
      int b = tid;
      float dot = 0.0f;
#pragma unroll 8
      for (int j = 0; j < A_DIM; ++j) dot += a_l[b * A_DIM + j] * WA3[j];
      float nm = 0.2f * (dot + bA3[0]) + 0.5f;          // hard sigmoid
      nm = fminf(fmaxf(nm, 0.0f), 1.0f);
      float lv = (t == S_LEN - 1) ? 1.0f : 0.0f;
      float pm, phv;
      if (level == 0) {
        pm = maskBS[b * S_LEN + t];
        phv = pm;
      } else {
        pm  = (t < S_LEN - 1) ? NMSp[(t + 1) * BATCH + b] : 1.0f;  // shifted nms
        phv = HVSp[t * BATCH + b];
      }
      nm = (1.0f - lv) * (pm * phv * nm);
      float hvp  = hv_l[b];
      float both = pm * phv * (1.0f - nm) * hvp;
      float xo   = pm * phv * (nm + (1.0f - nm) * (1.0f - hvp));
      float ho   = (1.0f - pm + pm * (1.0f - phv)) * (1.0f - nm) * hvp;
      float hv   = both + xo + ho;
      both_l[b] = both; xo_l[b] = xo; ho_l[b] = ho; hv_l[b] = hv;
      NMSc[t * BATCH + b] = nm;
      HVSc[t * BATCH + b] = hv;
    }
    __syncthreads();

    // ---- Phase 3: h = both*lrelu(h_@W1 + b1) + xo*xt + ho*h_prev ----
    for (int tt = wave; tt < 64; tt += 32) {            // 4 x 16 tiles
      int tm = tt >> 4, tn = tt & 15;
      v8f acc = {};
#pragma unroll
      for (int kp = 0; kp < 16; ++kp) {
        v16h av = ld_pack(hsA, tm * 16 + kp, lane);
        v16h bv = ld_pack(W1p, kp * 16 + tn, lane);
        acc = wmma_f16f32(av, bv, acc);
      }
      int col = tn * 16 + lm;
      int kp = col >> 5, kk = col & 31;
      float b1c = b1[col];
#pragma unroll
      for (int r2 = 0; r2 < 8; ++r2) {
        int m = r2 + mo;
        int row = tm * 16 + m;
        float hW = lrelu(acc[r2] + b1c);
        float xt = XE_t[(size_t)row * H_DIM + col];
        float hp = h_l[row * H_DIM + col];
        float hn = both_l[row] * hW + xo_l[row] * xt + ho_l[row] * hp;
        HS[(size_t)t * BATCH * H_DIM + (size_t)row * H_DIM + col] = hn;
        h_l[row * H_DIM + col] = hn;
        hA[a_pack_idx(tm, 8, kp, m, kk)] = (_Float16)hn;   // next step's A operand
        if (lastLevel && t == S_LEN - 1) OUT[(size_t)row * H_DIM + col] = hn;
      }
    }
    __syncthreads();
  }
}

// ---------------- host orchestration ----------------
extern "C" void kernel_launch(void* const* d_in, const int* in_sizes, int n_in,
                              void* d_out, int out_size, void* d_ws, size_t ws_size,
                              hipStream_t stream) {
  const float* x     = (const float*)d_in[0];   // [B,S,IN]
  const float* mask  = (const float*)d_in[1];   // [B,S]
  /* d_in[2] = bucket_size (compile-time S_LEN) */
  const float* W_emb = (const float*)d_in[3];
  const float* b_emb = (const float*)d_in[4];
  const float* W     = (const float*)d_in[5];
  const float* U     = (const float*)d_in[6];
  const float* b     = (const float*)d_in[7];
  const float* W1    = (const float*)d_in[8];
  const float* b1    = (const float*)d_in[9];
  const float* WA1   = (const float*)d_in[10];
  const float* UA1   = (const float*)d_in[11];
  const float* bA1   = (const float*)d_in[12];
  const float* WA3   = (const float*)d_in[13];
  const float* bA3   = (const float*)d_in[14];
  float* out = (float*)d_out;

  // workspace carve (256B aligned blocks)
  char* base = (char*)d_ws;
  size_t off = 0;
  auto alloc = [&](size_t bytes) -> void* {
    void* p = base + off;
    off += (bytes + 255) & ~(size_t)255;
    return p;
  };
  float* XE0 = (float*)alloc((size_t)S_LEN * BATCH * H_DIM * 4);
  float* XE1 = (float*)alloc((size_t)S_LEN * BATCH * H_DIM * 4);
  float* XWs = (float*)alloc((size_t)S_LEN * BATCH * INNER * 4);
  float* XAs = (float*)alloc((size_t)S_LEN * BATCH * A_DIM * 4);
  float* NMS0 = (float*)alloc((size_t)S_LEN * BATCH * 4);
  float* NMS1 = (float*)alloc((size_t)S_LEN * BATCH * 4);
  float* HVS0 = (float*)alloc((size_t)S_LEN * BATCH * 4);
  float* HVS1 = (float*)alloc((size_t)S_LEN * BATCH * 4);
  _Float16* W_emb_p = (_Float16*)alloc((size_t)IN_DIM * H_DIM * 2);
  _Float16* W_p     = (_Float16*)alloc((size_t)H_DIM * INNER * 2);
  _Float16* U_p     = (_Float16*)alloc((size_t)H_DIM * INNER * 2);
  _Float16* W1_p    = (_Float16*)alloc((size_t)INNER * H_DIM * 2);
  _Float16* WA1_p   = (_Float16*)alloc((size_t)H_DIM * A_DIM * 2);
  _Float16* UA1_p   = (_Float16*)alloc((size_t)H_DIM * A_DIM * 2);
  (void)ws_size; (void)n_in; (void)in_sizes; (void)out_size;

  // pack weights into WMMA-B lane layout (f16)
  pack_b<<<(IN_DIM * H_DIM + 255) / 256, 256, 0, stream>>>(W_emb, W_emb_p, IN_DIM, H_DIM);
  pack_b<<<(H_DIM * INNER + 255) / 256, 256, 0, stream>>>(W, W_p, H_DIM, INNER);
  pack_b<<<(H_DIM * INNER + 255) / 256, 256, 0, stream>>>(U, U_p, H_DIM, INNER);
  pack_b<<<(INNER * H_DIM + 255) / 256, 256, 0, stream>>>(W1, W1_p, INNER, H_DIM);
  pack_b<<<(H_DIM * A_DIM + 255) / 256, 256, 0, stream>>>(WA1, WA1_p, H_DIM, A_DIM);
  pack_b<<<(H_DIM * A_DIM + 255) / 256, 256, 0, stream>>>(UA1, UA1_p, H_DIM, A_DIM);

  // embedding: XE0[s*B+b,:] = x[b,s,:] @ W_emb + b_emb   (M=16384, N=256, K=256)
  {
    int tiles = (S_LEN * BATCH / 16) * (H_DIM / 64);
    gemm_wmma<<<tiles / 8, 256, 0, stream>>>(x, IN_DIM, 1, W_emb_p, b_emb, XE0,
                                             S_LEN * BATCH, H_DIM, IN_DIM);
  }

  (void)hipFuncSetAttribute((const void*)scan_level,
                            hipFuncAttributeMaxDynamicSharedMemorySize,
                            SCAN_LDS_BYTES);

  for (int lvl = 0; lvl < NLEVELS; ++lvl) {
    float* XEc = (lvl & 1) ? XE1 : XE0;
    float* XEn = (lvl & 1) ? XE0 : XE1;
    float* NMSc = (lvl & 1) ? NMS1 : NMS0;
    float* NMSp = (lvl & 1) ? NMS0 : NMS1;
    float* HVSc = (lvl & 1) ? HVS1 : HVS0;
    float* HVSp = (lvl & 1) ? HVS0 : HVS1;

    // time-parallel precompute: XW = XEc@W + b ; XA = XEc@W_action_1 + b_action_1
    {
      int tiles = (S_LEN * BATCH / 16) * (INNER / 64);
      gemm_wmma<<<tiles / 8, 256, 0, stream>>>(XEc, H_DIM, 0, W_p, b, XWs,
                                               S_LEN * BATCH, INNER, H_DIM);
    }
    {
      int tiles = (S_LEN * BATCH / 16) * (A_DIM / 64);
      gemm_wmma<<<tiles / 8, 256, 0, stream>>>(XEc, H_DIM, 0, WA1_p, bA1, XAs,
                                               S_LEN * BATCH, A_DIM, H_DIM);
    }
    // sequential recurrence: one persistent workgroup, state in LDS
    scan_level<<<1, 1024, SCAN_LDS_BYTES, stream>>>(
        XEc, XWs, XAs, U_p, W1_p, UA1_p, b1, WA3, bA3, mask,
        NMSp, HVSp, NMSc, HVSc, XEn, out, lvl, (lvl == NLEVELS - 1) ? 1 : 0);
  }
}
